// Qwen3_5VisionModel_25168508354987
// MI455X (gfx1250) — matrix operand order repaired and checked by reading, not compiled, not run on hardware
//
#include <hip/hip_runtime.h>
#include <hip/hip_bf16.h>
#include <math.h>

// ---------------------------------------------------------------------------
// CDNA5 (gfx1250) wave32 WMMA types
// ---------------------------------------------------------------------------
typedef __attribute__((ext_vector_type(16))) __bf16 bf16x16;
typedef __attribute__((ext_vector_type(8)))  __bf16 bf16x8;
typedef __attribute__((ext_vector_type(8)))  float  f32x8;

__device__ __forceinline__ f32x8 wmma_bf16(bf16x16 a, bf16x16 b, f32x8 c) {
  // D = A(16x32) * B(32x16) + C, f32 accumulate
  return __builtin_amdgcn_wmma_f32_16x16x32_bf16(false, a, false, b,
                                                 (short)0, c, false, false);
}

// A-fragment (16-bit A 16x32, contiguous k order; matches B-fragments here):
// lane l: m = l&15, h = l>>4; elems e -> K = 16h + e
// base must point at A[row*Kp + k0 + 16*h]
__device__ __forceinline__ bf16x16 load_a_frag(const __bf16* base) {
  bf16x8 lo = *(const bf16x8*)base;
  bf16x8 hi = *(const bf16x8*)(base + 8);
  bf16x16 a;
#pragma unroll
  for (int e = 0; e < 8; ++e) { a[e] = lo[e]; a[8 + e] = hi[e]; }
  return a;
}

// B-fragment (16-bit B 32x16): lane l: n = l&15, h = l>>4; elems e -> K = 16h + e
// base must point at 16 consecutive bf16 (k0 + 16h)
__device__ __forceinline__ bf16x16 load_b_frag_bf16(const __bf16* base) {
  bf16x8 lo = *(const bf16x8*)base;
  bf16x8 hi = *(const bf16x8*)(base + 8);
  bf16x16 b;
#pragma unroll
  for (int e = 0; e < 8; ++e) { b[e] = lo[e]; b[8 + e] = hi[e]; }
  return b;
}

// 16 consecutive f32 -> bf16x16 B-fragment (in-register weight conversion)
__device__ __forceinline__ bf16x16 load_b_frag_f32(const float* wp) {
  float t[16];
  *(float4*)(t + 0)  = ((const float4*)wp)[0];
  *(float4*)(t + 4)  = ((const float4*)wp)[1];
  *(float4*)(t + 8)  = ((const float4*)wp)[2];
  *(float4*)(t + 12) = ((const float4*)wp)[3];
  bf16x16 b;
#pragma unroll
  for (int e = 0; e < 16; ++e) b[e] = (__bf16)t[e];
  return b;
}

// ---------------------------------------------------------------------------
// Generic GEMM: C[M,N] = act(A_bf16[M,Kp] * W_f32[N,K]^T + bias) (+ residual)
// Weights are streamed f32 and converted to bf16 in-register (each weight is
// consumed once per GEMM; weights/activations are reused across blocks via L2).
// Workgroup = 8 waves, WG tile 64(M) x 128(N); each wave owns 32x32.
// M multiple of 64, N multiple of 128 (true for all call sites).
// Main K-loop is branch-free (K & ~31); the K tail (only patch-embed, K=1176)
// is handled once after the loop so no EXEC divergence pollutes the hot loop.
// act: 0=none, 1=tanh-GELU, 2=erf-GELU
// ---------------------------------------------------------------------------
__global__ __launch_bounds__(256)
void gemm_bf16_kernel(const __bf16* __restrict__ A, int Kp,
                      const float* __restrict__ W,
                      const float* __restrict__ bias,
                      const float* residual,
                      float* Cf, __bf16* Cb,
                      int M, int N, int K, int act)
{
  const int lane = threadIdx.x & 31;
  const int wave = threadIdx.x >> 5;
  const int lm = lane & 15, lh = lane >> 4;
  const int m0 = blockIdx.y * 64 + (wave & 1) * 32;
  const int n0 = blockIdx.x * 128 + (wave >> 1) * 32;

  const __bf16* a0 = A + (size_t)(m0 + lm) * Kp + 16 * lh;
  const __bf16* a1 = A + (size_t)(m0 + 16 + lm) * Kp + 16 * lh;
  const float*  w0 = W + (size_t)(n0 + lm) * K + 16 * lh;
  const float*  w1 = W + (size_t)(n0 + 16 + lm) * K + 16 * lh;

  f32x8 acc[2][2] = {};

  const int Kmain = K & ~31;
  for (int k0 = 0; k0 < Kmain; k0 += 32) {
    bf16x16 af0 = load_a_frag(a0 + k0);
    bf16x16 af1 = load_a_frag(a1 + k0);
    bf16x16 bf0 = load_b_frag_f32(w0 + k0);
    bf16x16 bf1 = load_b_frag_f32(w1 + k0);
    acc[0][0] = wmma_bf16(af0, bf0, acc[0][0]);
    acc[0][1] = wmma_bf16(af0, bf1, acc[0][1]);
    acc[1][0] = wmma_bf16(af1, bf0, acc[1][0]);
    acc[1][1] = wmma_bf16(af1, bf1, acc[1][1]);
  }

  if (Kmain < K) { // executed once; A is zero-padded to Kp so A loads are safe
    bf16x16 af0 = load_a_frag(a0 + Kmain);
    bf16x16 af1 = load_a_frag(a1 + Kmain);
    const int kb = Kmain + 16 * lh;
    bf16x16 bf0, bf1;
#pragma unroll
    for (int e = 0; e < 16; ++e) {
      const bool ok = (kb + e) < K;
      bf0[e] = ok ? (__bf16)w0[Kmain + e] : (__bf16)0.0f;
      bf1[e] = ok ? (__bf16)w1[Kmain + e] : (__bf16)0.0f;
    }
    acc[0][0] = wmma_bf16(af0, bf0, acc[0][0]);
    acc[0][1] = wmma_bf16(af0, bf1, acc[0][1]);
    acc[1][0] = wmma_bf16(af1, bf0, acc[1][0]);
    acc[1][1] = wmma_bf16(af1, bf1, acc[1][1]);
  }

  // Epilogue. C/D layout: VGPR r -> row = 16i + 8h + r, lane -> col = 16j + (l&15)
#pragma unroll
  for (int i = 0; i < 2; ++i) {
#pragma unroll
    for (int j = 0; j < 2; ++j) {
      const int col = n0 + 16 * j + lm;
      const float bs = bias ? bias[col] : 0.0f;
#pragma unroll
      for (int r = 0; r < 8; ++r) {
        const int row = m0 + 16 * i + 8 * lh + r;
        float v = acc[i][j][r] + bs;
        if (act == 1) {
          float u = 0.7978845608028654f * (v + 0.044715f * v * v * v);
          v = 0.5f * v * (1.0f + tanhf(u));
        } else if (act == 2) {
          v = 0.5f * v * (1.0f + erff(v * 0.7071067811865476f));
        }
        const size_t off = (size_t)row * N + col;
        if (residual) v += residual[off];
        if (Cf) Cf[off] = v;
        if (Cb) Cb[off] = (__bf16)v;
      }
    }
  }
}

// ---------------------------------------------------------------------------
// f32 -> bf16 with K padding (zeros) for WMMA A operands
// ---------------------------------------------------------------------------
__global__ void cvt_pad_kernel(const float* __restrict__ in, __bf16* __restrict__ out,
                               int K, int Kp, int total)
{
  int idx = blockIdx.x * blockDim.x + threadIdx.x;
  if (idx >= total) return;
  int r = idx / Kp, c = idx % Kp;
  out[idx] = (c < K) ? (__bf16)in[(size_t)r * K + c] : (__bf16)0.0f;
}

// x[n,c] += pos_table[pos_ids[n], c]
__global__ void posadd_kernel(float* __restrict__ x, const float* __restrict__ table,
                              const int* __restrict__ ids, int H, int total)
{
  int idx = blockIdx.x * blockDim.x + threadIdx.x;
  if (idx >= total) return;
  int row = idx / H, c = idx % H;
  x[idx] += table[(size_t)ids[row] * H + c];
}

// ---------------------------------------------------------------------------
// LayerNorm over last dim C, writes bf16 (feeds the next WMMA GEMM directly)
// one row per 256-thread block (8 waves)
// ---------------------------------------------------------------------------
__global__ __launch_bounds__(256)
void layernorm_kernel(const float* __restrict__ x, const float* __restrict__ w,
                      const float* __restrict__ b, __bf16* __restrict__ out, int C)
{
  const int row = blockIdx.x;
  const float* xr = x + (size_t)row * C;
  float s = 0.0f, s2 = 0.0f;
  for (int c = threadIdx.x; c < C; c += 256) { float v = xr[c]; s += v; s2 += v * v; }
#pragma unroll
  for (int o = 16; o; o >>= 1) { s += __shfl_down(s, o, 32); s2 += __shfl_down(s2, o, 32); }
  __shared__ float ps[8], ps2[8], stat[2];
  const int wv = threadIdx.x >> 5, ln = threadIdx.x & 31;
  if (!ln) { ps[wv] = s; ps2[wv] = s2; }
  __syncthreads();
  if (threadIdx.x == 0) {
    float S = 0.0f, S2 = 0.0f;
#pragma unroll
    for (int i = 0; i < 8; ++i) { S += ps[i]; S2 += ps2[i]; }
    float mean = S / C;
    float var  = S2 / C - mean * mean;
    stat[0] = mean;
    stat[1] = rsqrtf(var + 1e-6f);
  }
  __syncthreads();
  const float mean = stat[0], inv = stat[1];
  for (int c = threadIdx.x; c < C; c += 256)
    out[(size_t)row * C + c] = (__bf16)((xr[c] - mean) * inv * w[c] + b[c]);
}

// ---------------------------------------------------------------------------
// QKV split + RoPE -> bf16 WMMA-friendly layouts
//   Qb/Kb: [head][tok][96]  (head-dim zero-padded 80->96, multiple of 32 for K-loop)
//   Vt:    [head][dim80][tok]  (so P*V B-fragments are contiguous)
// ---------------------------------------------------------------------------
__global__ void rope_split_kernel(const float* __restrict__ qkv,
                                  const float* __restrict__ cosb,
                                  const float* __restrict__ sinb,
                                  __bf16* __restrict__ Qb, __bf16* __restrict__ Kb,
                                  __bf16* __restrict__ Vt, int Ntok)
{
  int idx = blockIdx.x * blockDim.x + threadIdx.x;
  const int total = Ntok * 16 * 96;
  if (idx >= total) return;
  const int d = idx % 96;
  const int t = idx / 96;
  const int h = t % 16;
  const int n = t / 16;
  const size_t qoff = ((size_t)h * Ntok + n) * 96 + d;
  if (d < 80) {
    const float* row = qkv + (size_t)n * 3840;
    const float qv = row[h * 80 + d];
    const float kv = row[1280 + h * 80 + d];
    const float vv = row[2560 + h * 80 + d];
    const float cs = cosb[n * 80 + d];
    const float sn = sinb[n * 80 + d];
    const float qr = (d < 40) ? -row[h * 80 + d + 40] : row[h * 80 + d - 40];
    const float kr = (d < 40) ? -row[1280 + h * 80 + d + 40] : row[1280 + h * 80 + d - 40];
    Qb[qoff] = (__bf16)(qv * cs + qr * sn);
    Kb[qoff] = (__bf16)(kv * cs + kr * sn);
    Vt[((size_t)h * 80 + d) * Ntok + n] = (__bf16)vv;
  } else {
    Qb[qoff] = (__bf16)0.0f;
    Kb[qoff] = (__bf16)0.0f;
  }
}

// ---------------------------------------------------------------------------
// Flash attention, one wave (32 threads) per (head, 32-query-row block).
// Segment mask handled by restricting the column loop to [cu[s], cu[s+1])
// (boundaries are 32-aligned for this workload).
// Scores: WMMA over padded head-dim 96 (3 K-steps). Online softmax with
// shfl_xor row reductions. P goes C-layout -> A-layout through a 2 KB LDS
// tile (single-wave LDS ops are in-order). Then P*V via WMMA (5 dim tiles).
// ---------------------------------------------------------------------------
__global__ __launch_bounds__(32)
void attn_kernel(const __bf16* __restrict__ Qb, const __bf16* __restrict__ Kb,
                 const __bf16* __restrict__ Vt, const int* __restrict__ cu,
                 int nseg, __bf16* __restrict__ O, int Ntok)
{
  __shared__ __bf16 pbuf[32 * 32];
  const int head = blockIdx.y;
  const int m0 = blockIdx.x * 32;
  const int lane = threadIdx.x & 31;
  const int lm = lane & 15, lh = lane >> 4;

  int s0 = 0, s1 = Ntok;
  for (int s = 0; s < nseg; ++s) {
    int a = cu[s], b = cu[s + 1];
    if (m0 >= a && m0 < b) { s0 = a; s1 = b; }
  }

  const float scale = 0.11180339887498949f; // 80^-0.5
  const __bf16* Qh = Qb + (size_t)head * Ntok * 96;
  const __bf16* Kh = Kb + (size_t)head * Ntok * 96;
  const __bf16* Vh = Vt + (size_t)head * 80 * Ntok;

  bf16x16 qf[2][3];
#pragma unroll
  for (int i = 0; i < 2; ++i)
#pragma unroll
    for (int ks = 0; ks < 3; ++ks)
      qf[i][ks] = load_a_frag(Qh + (size_t)(m0 + 16 * i + lm) * 96 + 32 * ks + 16 * lh);

  f32x8 acc[2][5] = {};
  float rmax[2][8], rsum[2][8];
#pragma unroll
  for (int i = 0; i < 2; ++i)
#pragma unroll
    for (int r = 0; r < 8; ++r) { rmax[i][r] = -1e30f; rsum[i][r] = 0.0f; }

  for (int c0 = s0; c0 < s1; c0 += 32) {
    f32x8 sc[2][2] = {};
#pragma unroll
    for (int ks = 0; ks < 3; ++ks) {
      bf16x16 kf[2];
#pragma unroll
      for (int j = 0; j < 2; ++j)
        kf[j] = load_b_frag_bf16(Kh + (size_t)(c0 + 16 * j + lm) * 96 + 32 * ks + 16 * lh);
#pragma unroll
      for (int i = 0; i < 2; ++i)
#pragma unroll
        for (int j = 0; j < 2; ++j)
          sc[i][j] = wmma_bf16(qf[i][ks], kf[j], sc[i][j]);
    }
    // online softmax (rows are shared by the 16 lanes with equal lh)
#pragma unroll
    for (int i = 0; i < 2; ++i) {
#pragma unroll
      for (int r = 0; r < 8; ++r) {
        float v0 = sc[i][0][r] * scale;
        float v1 = sc[i][1][r] * scale;
        float mx = fmaxf(v0, v1);
#pragma unroll
        for (int o = 8; o; o >>= 1) mx = fmaxf(mx, __shfl_xor(mx, o, 32));
        const float nm = fmaxf(rmax[i][r], mx);
        const float corr = __expf(rmax[i][r] - nm);
        rmax[i][r] = nm;
        const float p0 = __expf(v0 - nm), p1 = __expf(v1 - nm);
        float ps = p0 + p1;
#pragma unroll
        for (int o = 8; o; o >>= 1) ps += __shfl_xor(ps, o, 32);
        rsum[i][r] = rsum[i][r] * corr + ps;
#pragma unroll
        for (int j = 0; j < 5; ++j) acc[i][j][r] *= corr;
        sc[i][0][r] = p0; sc[i][1][r] = p1;
      }
    }
    // transpose P through LDS: C-layout -> A-layout
#pragma unroll
    for (int i = 0; i < 2; ++i)
#pragma unroll
      for (int jj = 0; jj < 2; ++jj)
#pragma unroll
        for (int r = 0; r < 8; ++r)
          pbuf[(16 * i + 8 * lh + r) * 32 + 16 * jj + lm] = (__bf16)sc[i][jj][r];
    bf16x16 pa[2];
#pragma unroll
    for (int i = 0; i < 2; ++i) {
      const int mrow = 16 * i + lm;
      bf16x16 a;
#pragma unroll
      for (int e = 0; e < 8; ++e) {
        a[e]     = pbuf[mrow * 32 + 16 * lh + e];
        a[8 + e] = pbuf[mrow * 32 + 16 * lh + 8 + e];
      }
      pa[i] = a;
    }
    // P (32 cols = one K-step) * V
#pragma unroll
    for (int j = 0; j < 5; ++j) {
      bf16x16 bv = load_b_frag_bf16(Vh + (size_t)(16 * j + lm) * Ntok + c0 + 16 * lh);
#pragma unroll
      for (int i = 0; i < 2; ++i)
        acc[i][j] = wmma_bf16(pa[i], bv, acc[i][j]);
    }
  }

#pragma unroll
  for (int i = 0; i < 2; ++i)
#pragma unroll
    for (int j = 0; j < 5; ++j)
#pragma unroll
      for (int r = 0; r < 8; ++r) {
        const int row = m0 + 16 * i + 8 * lh + r;
        O[(size_t)row * 1280 + head * 80 + 16 * j + lm] =
            (__bf16)(acc[i][j][r] / rsum[i][r]);
      }
}

// ---------------------------------------------------------------------------
// Host orchestration
// ---------------------------------------------------------------------------
extern "C" void kernel_launch(void* const* d_in, const int* in_sizes, int n_in,
                              void* d_out, int out_size, void* d_ws, size_t ws_size,
                              hipStream_t stream)
{
  (void)in_sizes; (void)n_in; (void)out_size; (void)ws_size;
  const float* hidden    = (const float*)d_in[0];
  const int*   cu        = (const int*)d_in[1];
  const int*   pos_ids   = (const int*)d_in[2];
  const float* cosb      = (const float*)d_in[3];
  const float* sinb      = (const float*)d_in[4];
  const float* pe_w      = (const float*)d_in[5];
  const float* pe_b      = (const float*)d_in[6];
  const float* pos_table = (const float*)d_in[7];
  const float* ln1_w     = (const float*)d_in[8];
  const float* ln1_b     = (const float*)d_in[9];
  const float* ln2_w     = (const float*)d_in[10];
  const float* ln2_b     = (const float*)d_in[11];
  const float* qkv_w     = (const float*)d_in[12];
  const float* qkv_b     = (const float*)d_in[13];
  const float* proj_w    = (const float*)d_in[14];
  const float* proj_b    = (const float*)d_in[15];
  const float* fc1_w     = (const float*)d_in[16];
  const float* fc1_b     = (const float*)d_in[17];
  const float* fc2_w     = (const float*)d_in[18];
  const float* fc2_b     = (const float*)d_in[19];
  const float* mnw       = (const float*)d_in[20];
  const float* mnb       = (const float*)d_in[21];
  const float* mfc1_w    = (const float*)d_in[22];
  const float* mfc1_b    = (const float*)d_in[23];
  const float* mfc2_w    = (const float*)d_in[24];
  const float* mfc2_b    = (const float*)d_in[25];

  const int N = 2048, H = 1280, IN = 1176, INP = 1184, FF = 3456, HEADS = 16, HDP = 96;

  char* p = (char*)d_ws;
  auto carve = [&](size_t bytes) { void* r = (void*)p; p += (bytes + 255) & ~(size_t)255; return r; };
  __bf16* hsbf = (__bf16*)carve((size_t)N * INP * 2);
  float*  x    = (float*) carve((size_t)N * H * 4);
  __bf16* hbf  = (__bf16*)carve((size_t)N * H * 2);
  float*  qkvf = (float*) carve((size_t)N * 3840 * 4);
  __bf16* Qbuf = (__bf16*)carve((size_t)HEADS * N * HDP * 2);
  __bf16* Kbuf = (__bf16*)carve((size_t)HEADS * N * HDP * 2);
  __bf16* Vt   = (__bf16*)carve((size_t)HEADS * 80 * N * 2);
  __bf16* obf  = (__bf16*)carve((size_t)N * H * 2);
  __bf16* f1bf = (__bf16*)carve((size_t)N * FF * 2);
  __bf16* ybf  = (__bf16*)carve((size_t)N * H * 2);
  __bf16* m1bf = (__bf16*)carve((size_t)512 * 5120 * 2);

  // patch embed: convert/pad input once (activations are reused across N tiles)
  {
    const int total = N * INP;
    cvt_pad_kernel<<<(total + 255) / 256, 256, 0, stream>>>(hidden, hsbf, IN, INP, total);
  }
  gemm_bf16_kernel<<<dim3(H / 128, N / 64), 256, 0, stream>>>(
      hsbf, INP, pe_w, pe_b, nullptr, x, nullptr, N, H, IN, 0);
  {
    const int total = N * H;
    posadd_kernel<<<(total + 255) / 256, 256, 0, stream>>>(x, pos_table, pos_ids, H, total);
  }

  for (int l = 0; l < 2; ++l) {
    layernorm_kernel<<<N, 256, 0, stream>>>(x, ln1_w + l * H, ln1_b + l * H, hbf, H);
    gemm_bf16_kernel<<<dim3(3840 / 128, N / 64), 256, 0, stream>>>(
        hbf, H, qkv_w + (size_t)l * 3840 * H, qkv_b + l * 3840,
        nullptr, qkvf, nullptr, N, 3840, H, 0);
    {
      const int total = N * HEADS * HDP;
      rope_split_kernel<<<(total + 255) / 256, 256, 0, stream>>>(
          qkvf, cosb, sinb, Qbuf, Kbuf, Vt, N);
    }
    attn_kernel<<<dim3(N / 32, HEADS), 32, 0, stream>>>(Qbuf, Kbuf, Vt, cu, 2, obf, N);
    gemm_bf16_kernel<<<dim3(H / 128, N / 64), 256, 0, stream>>>(
        obf, H, proj_w + (size_t)l * H * H, proj_b + l * H,
        x, x, nullptr, N, H, H, 0);
    layernorm_kernel<<<N, 256, 0, stream>>>(x, ln2_w + l * H, ln2_b + l * H, hbf, H);
    gemm_bf16_kernel<<<dim3(FF / 128, N / 64), 256, 0, stream>>>(
        hbf, H, fc1_w + (size_t)l * FF * H, fc1_b + l * FF,
        nullptr, nullptr, f1bf, N, FF, H, 1);
    gemm_bf16_kernel<<<dim3(H / 128, N / 64), 256, 0, stream>>>(
        f1bf, FF, fc2_w + (size_t)l * H * FF, fc2_b + l * H,
        x, x, nullptr, N, H, FF, 0);
  }

  // merger: LN -> view [512, 5120] (contiguous) -> GEMM(erf-gelu) -> GEMM
  layernorm_kernel<<<N, 256, 0, stream>>>(x, mnw, mnb, ybf, H);
  gemm_bf16_kernel<<<dim3(5120 / 128, 512 / 64), 256, 0, stream>>>(
      ybf, 5120, mfc1_w, mfc1_b, nullptr, nullptr, m1bf, 512, 5120, 5120, 2);
  gemm_bf16_kernel<<<dim3(2048 / 128, 512 / 64), 256, 0, stream>>>(
      m1bf, 5120, mfc2_w, mfc2_b, nullptr, (float*)d_out, nullptr, 512, 2048, 5120, 0);
}
